// attention_69148973465717
// MI455X (gfx1250) — matrix operand order repaired and checked by reading, not compiled, not run on hardware
//
#include <hip/hip_runtime.h>

typedef __attribute__((ext_vector_type(16))) __bf16 v16bf;
typedef __attribute__((ext_vector_type(8)))  __bf16 v8bf;
typedef __attribute__((ext_vector_type(8)))  float  v8f;

union ABf { v16bf v; v8bf h[2]; };

#define SEQ    2048
#define DMODEL 2048
#define NHEAD  16
#define HDIM   128

// ---------------------------------------------------------------------------
// block tree reduction (power-of-2 n, all threads participate)
// ---------------------------------------------------------------------------
__device__ __forceinline__ float blk_reduce_add(float v, float* red, int i, int n) {
  red[i] = v;
  __syncthreads();
  for (int off = n >> 1; off > 0; off >>= 1) {
    if (i < off) red[i] += red[i + off];
    __syncthreads();
  }
  float r = red[0];
  __syncthreads();
  return r;
}

// ---------------------------------------------------------------------------
// Kernel 1: dual LayerNorm of x rows -> bf16, plus per-token L2 norm m[c]
// ---------------------------------------------------------------------------
__global__ void ln_x_kernel(const float* __restrict__ x,
                            const float* __restrict__ qg, const float* __restrict__ qb,
                            const float* __restrict__ kg, const float* __restrict__ kb_,
                            __bf16* __restrict__ Xq, __bf16* __restrict__ Xkv,
                            float* __restrict__ mout) {
  __shared__ float red[256];
  const int c = blockIdx.x, t = threadIdx.x;
  const float* row = x + (size_t)c * DMODEL;
  float vals[8], s = 0.f, ss = 0.f;
#pragma unroll
  for (int j = 0; j < 8; ++j) {
    float v = row[t + j * 256];
    vals[j] = v; s += v; ss += v * v;
  }
  float sum   = blk_reduce_add(s,  red, t, 256);
  float sumsq = blk_reduce_add(ss, red, t, 256);
  const float mu = sum * (1.f / 2048.f);
  const float rs = rsqrtf(sumsq * (1.f / 2048.f) - mu * mu + 1e-5f);
#pragma unroll
  for (int j = 0; j < 8; ++j) {
    const int idx = t + j * 256;
    const float xn = (vals[j] - mu) * rs;
    Xq [(size_t)c * DMODEL + idx] = (__bf16)(xn * qg[idx] + qb[idx]);
    Xkv[(size_t)c * DMODEL + idx] = (__bf16)(xn * kg[idx] + kb_[idx]);
  }
  if (t == 0) mout[c] = sqrtf(sumsq);
}

// ---------------------------------------------------------------------------
// Kernel 2: transpose f32 [R][Cc] -> bf16 [Cc][R]  (32x32 LDS tile)
// ---------------------------------------------------------------------------
__global__ void transpose_kernel(const float* __restrict__ in, __bf16* __restrict__ out,
                                 int R, int Cc) {
  __shared__ float tile[32][33];
  const int bx = blockIdx.x * 32, by = blockIdx.y * 32;
  const int tx = threadIdx.x, ty = threadIdx.y;  // (32,8)
#pragma unroll
  for (int i = 0; i < 32; i += 8)
    tile[ty + i][tx] = in[(size_t)(by + ty + i) * Cc + bx + tx];
  __syncthreads();
#pragma unroll
  for (int i = 0; i < 32; i += 8)
    out[(size_t)(bx + ty + i) * R + by + tx] = (__bf16)tile[tx][ty + i];
}

// ---------------------------------------------------------------------------
// Kernel 3: GEMM  out[M][N](f32) = A[M][K](bf16) @ Bt[N][K]^T(bf16) + bias
// wave = 32(M) x 64(N): 2 A-frags x 4 B-frags -> 8 WMMAs per 32-K step.
// K is a template constant so ALL intra-tile displacements become 24-bit
// immediate offsets: the hot loop advances exactly two pointers by 64B.
// block = 256 threads = 8 waves as (2M x 4N) -> block tile 64M x 256N
// ---------------------------------------------------------------------------
template <int K>
__global__ void gemm_bf16_kernel(const __bf16* __restrict__ A,
                                 const __bf16* __restrict__ Bt,
                                 const float* __restrict__ bias,
                                 float* __restrict__ out,
                                 int M, int N) {
  const int lane = threadIdx.x & 31, wv = threadIdx.x >> 5;
  const int wm = wv >> 2;                       // 0..1
  const int wn = wv & 3;                        // 0..3
  const int m = lane & 15, half = lane >> 4;
  const int m0 = blockIdx.y * 64 + wm * 32;
  const int n0 = blockIdx.x * 256 + wn * 64;
  const v8f vz = {0.f, 0.f, 0.f, 0.f, 0.f, 0.f, 0.f, 0.f};
  v8f c[2][4];
#pragma unroll
  for (int mi = 0; mi < 2; ++mi)
#pragma unroll
    for (int t = 0; t < 4; ++t) c[mi][t] = vz;
  const __bf16* aptr = A  + (size_t)(m0 + m) * K + 16 * half;
  const __bf16* bptr = Bt + (size_t)(n0 + m) * K + 16 * half;
  for (int k0 = 0; k0 < K; k0 += 32) {
    ABf a0, a1;
    a0.h[0] = *(const v8bf*)(aptr);                 // K = 16*half + e
    a0.h[1] = *(const v8bf*)(aptr + 8);             // K = 16*half + 8 + e
    a1.h[0] = *(const v8bf*)(aptr + 16 * K);        // row m0+16+m, imm offset
    a1.h[1] = *(const v8bf*)(aptr + 16 * K + 8);
#pragma unroll
    for (int t = 0; t < 4; ++t) {
      ABf b;
      b.v = *(const v16bf*)(bptr + t * (16 * K));   // imm offset per n-tile
      c[0][t] = __builtin_amdgcn_wmma_f32_16x16x32_bf16(false, a0.v, false, b.v,
                                                        (short)0, c[0][t], false, false);
      c[1][t] = __builtin_amdgcn_wmma_f32_16x16x32_bf16(false, a1.v, false, b.v,
                                                        (short)0, c[1][t], false, false);
    }
    aptr += 32;
    bptr += 32;
  }
#pragma unroll
  for (int t = 0; t < 4; ++t) {
    const int col = n0 + t * 16 + m;
    const float bv = bias[col];
#pragma unroll
    for (int mi = 0; mi < 2; ++mi)
#pragma unroll
      for (int r = 0; r < 8; ++r)
        out[(size_t)(m0 + mi * 16 + r + 8 * half) * N + col] = c[mi][t][r] + bv;
  }
}

// ---------------------------------------------------------------------------
// Kernel 4: rotary (polar, magnitude = ||x||) + per-head LayerNorm
// writes qh [H][C][128] (x 1/sqrt(hd)), kh [H][C][128], vT [H][128][C]  (bf16)
// block = 64 threads (one per even/odd pair), grid = (C, H)
// ---------------------------------------------------------------------------
__global__ void rope_headln_kernel(const float* __restrict__ qp,
                                   const float* __restrict__ kvp,
                                   const float* __restrict__ mn,
                                   const float* __restrict__ ln_g,
                                   const float* __restrict__ ln_b,
                                   __bf16* __restrict__ qh,
                                   __bf16* __restrict__ kh,
                                   __bf16* __restrict__ vT) {
  __shared__ float red[64];
  const int c = blockIdx.x, h = blockIdx.y, i = threadIdx.x;
  const int d0 = 2 * i, d1 = d0 + 1;
  const float scale = 0.29730177875068026f;        // 128^-0.25
  const float mnv = mn[c];
  const float ang = (float)c * ((float)d0 * (10.308952660644293f / 128.0f)); // log(30000)/D
  float sa, ca;
  __sincosf(ang, &sa, &ca);
  const float fre = mnv * ca, fim = mnv * sa;
  const size_t qbase = (size_t)c * DMODEL + h * HDIM;
  const size_t kbase = (size_t)c * (2 * DMODEL) + h * HDIM;
  float a = qp[qbase + d0] * scale, b = qp[qbase + d1] * scale;
  const float q0 = a * fre - b * fim, q1 = a * fim + b * fre;
  a = kvp[kbase + d0] * scale; b = kvp[kbase + d1] * scale;
  const float k0 = a * fre - b * fim, k1 = a * fim + b * fre;
  const float qs  = blk_reduce_add(q0 + q1,           red, i, 64);
  const float qss = blk_reduce_add(q0 * q0 + q1 * q1, red, i, 64);
  const float ks  = blk_reduce_add(k0 + k1,           red, i, 64);
  const float kss = blk_reduce_add(k0 * k0 + k1 * k1, red, i, 64);
  const float qmu = qs * (1.f / 128.f);
  const float qrs = rsqrtf(qss * (1.f / 128.f) - qmu * qmu + 1e-5f);
  const float kmu = ks * (1.f / 128.f);
  const float krs = rsqrtf(kss * (1.f / 128.f) - kmu * kmu + 1e-5f);
  const float g0 = ln_g[d0], g1 = ln_g[d1], bb0 = ln_b[d0], bb1 = ln_b[d1];
  const float sm = 0.08838834764831845f;           // 1/sqrt(128) folded into q
  const size_t ho = ((size_t)h * SEQ + c) * HDIM;
  qh[ho + d0] = (__bf16)(((q0 - qmu) * qrs * g0 + bb0) * sm);
  qh[ho + d1] = (__bf16)(((q1 - qmu) * qrs * g1 + bb1) * sm);
  kh[ho + d0] = (__bf16)((k0 - kmu) * krs * g0 + bb0);
  kh[ho + d1] = (__bf16)((k1 - kmu) * krs * g1 + bb1);
  const size_t vbase = (size_t)c * (2 * DMODEL) + DMODEL + h * HDIM;
  vT[((size_t)h * HDIM + d0) * SEQ + c] = (__bf16)kvp[vbase + d0];
  vT[((size_t)h * HDIM + d1) * SEQ + c] = (__bf16)kvp[vbase + d1];
}

// ---------------------------------------------------------------------------
// Kernel 5: flash attention, causal. wave = 16 queries x hd=128.
// Online softmax; P -> A-fragment via per-wave LDS staging.
// __launch_bounds__(256,1): large live set (O 64 + Q 32 + S 16 + stats) must
// stay in VGPRs -- default occupancy target caused scratch spills in the loop.
// block = 256 (8 waves), grid = NHEAD * (SEQ/16) / 8
// ---------------------------------------------------------------------------
__global__ void __launch_bounds__(256, 1)
flash_attn_kernel(const __bf16* __restrict__ qh,
                  const __bf16* __restrict__ kh,
                  const __bf16* __restrict__ vT,
                  __bf16* __restrict__ attn) {
  __shared__ __bf16 lds_p[8][16][32];
  const int lane = threadIdx.x & 31;
  const int wv   = threadIdx.x >> 5;
  const int w    = blockIdx.x * 8 + wv;
  const int h    = w >> 7;               // 128 q-tiles per head
  const int q0   = (w & 127) << 4;
  const int m    = lane & 15;
  const int half = lane >> 4;

  // Q A-fragments for the whole hd=128 (4 x K32)
  ABf qf[4];
  const __bf16* qrow = qh + ((size_t)h * SEQ + (q0 + m)) * HDIM + 16 * half;
#pragma unroll
  for (int dt = 0; dt < 4; ++dt) {
    qf[dt].h[0] = *(const v8bf*)(qrow + dt * 32);
    qf[dt].h[1] = *(const v8bf*)(qrow + dt * 32 + 8);
  }

  const v8f vz = {0.f, 0.f, 0.f, 0.f, 0.f, 0.f, 0.f, 0.f};
  v8f o[8];
#pragma unroll
  for (int dt = 0; dt < 8; ++dt) o[dt] = vz;
  float rmax[8], rsum[8];
#pragma unroll
  for (int r = 0; r < 8; ++r) { rmax[r] = -3e38f; rsum[r] = 0.f; }

  // advancing base pointers (strides compile-time)
  const __bf16* kptr = kh + ((size_t)h * SEQ + m) * HDIM + 16 * half;
  const __bf16* vptr = vT + ((size_t)h * HDIM + m) * SEQ + 16 * half;

  for (int kb = 0; kb < q0 + 16; kb += 32) {
    // S = q @ k^T  (two 16x16 score tiles over this 32-wide k block)
    v8f s0 = vz, s1 = vz;
#pragma unroll
    for (int dt = 0; dt < 4; ++dt) {
      ABf b0, b1;
      b0.v = *(const v16bf*)(kptr + dt * 32);                  // rows kb+m
      b1.v = *(const v16bf*)(kptr + 16 * HDIM + dt * 32);      // rows kb+16+m
      s0 = __builtin_amdgcn_wmma_f32_16x16x32_bf16(false, qf[dt].v, false, b0.v,
                                                   (short)0, s0, false, false);
      s1 = __builtin_amdgcn_wmma_f32_16x16x32_bf16(false, qf[dt].v, false, b1.v,
                                                   (short)0, s1, false, false);
    }
    const int c0 = kb + m, c1 = kb + 16 + m;
#pragma unroll
    for (int r = 0; r < 8; ++r) {
      const int row = q0 + r + 8 * half;
      const bool ok0 = (c0 <= row), ok1 = (c1 <= row);
      float t = fmaxf(ok0 ? s0[r] : -3e38f, ok1 ? s1[r] : -3e38f);
#pragma unroll
      for (int off = 1; off < 16; off <<= 1) t = fmaxf(t, __shfl_xor(t, off, 32));
      const float nm   = fmaxf(rmax[r], t);
      const float corr = __expf(rmax[r] - nm);
      rmax[r] = nm;
      const float p0 = ok0 ? __expf(s0[r] - nm) : 0.f;
      const float p1 = ok1 ? __expf(s1[r] - nm) : 0.f;
      float ps = p0 + p1;
#pragma unroll
      for (int off = 1; off < 16; off <<= 1) ps += __shfl_xor(ps, off, 32);
      rsum[r] = rsum[r] * corr + ps;
#pragma unroll
      for (int dt = 0; dt < 8; ++dt) o[dt][r] *= corr;
      lds_p[wv][r + 8 * half][m]      = (__bf16)p0;
      lds_p[wv][r + 8 * half][16 + m] = (__bf16)p1;
    }
    asm volatile("s_wait_dscnt 0" ::: "memory");  // LDS RAW: stores -> A-frag loads
    ABf pf;
    pf.h[0] = *(const v8bf*)(&lds_p[wv][m][16 * half]);
    pf.h[1] = *(const v8bf*)(&lds_p[wv][m][16 * half + 8]);
    // O += P @ V   (V B-fragments from transposed v, K contiguous)
#pragma unroll
    for (int dt = 0; dt < 8; ++dt) {
      ABf vf;
      vf.v = *(const v16bf*)(vptr + dt * (16 * SEQ));          // imm offset per d-tile
      o[dt] = __builtin_amdgcn_wmma_f32_16x16x32_bf16(false, pf.v, false, vf.v,
                                                      (short)0, o[dt], false, false);
    }
    asm volatile("s_wait_dscnt 0" ::: "memory");  // LDS WAR before next block's stores
    kptr += 32 * HDIM;
    vptr += 32;
  }
#pragma unroll
  for (int r = 0; r < 8; ++r) rsum[r] = 1.f / rsum[r];
#pragma unroll
  for (int dt = 0; dt < 8; ++dt)
#pragma unroll
    for (int r = 0; r < 8; ++r)
      attn[(size_t)(q0 + r + 8 * half) * DMODEL + h * HDIM + dt * 16 + m] =
          (__bf16)(o[dt][r] * rsum[r]);
}

// ---------------------------------------------------------------------------
extern "C" void kernel_launch(void* const* d_in, const int* in_sizes, int n_in,
                              void* d_out, int out_size, void* d_ws, size_t ws_size,
                              hipStream_t stream) {
  (void)in_sizes; (void)n_in; (void)out_size; (void)ws_size;
  const float* x     = (const float*)d_in[0];
  const float* qn_g  = (const float*)d_in[1];
  const float* qn_b  = (const float*)d_in[2];
  const float* kvn_g = (const float*)d_in[3];
  const float* kvn_b = (const float*)d_in[4];
  const float* Wq    = (const float*)d_in[5];
  const float* bq    = (const float*)d_in[6];
  const float* Wkv   = (const float*)d_in[7];
  const float* bkv   = (const float*)d_in[8];
  const float* Wo    = (const float*)d_in[9];
  const float* bo    = (const float*)d_in[10];
  const float* ln_g  = (const float*)d_in[11];
  const float* ln_b  = (const float*)d_in[12];

  char* ws = (char*)d_ws;
  const size_t MB = 1024 * 1024;
  __bf16* Xq   = (__bf16*)(ws + 0);        // 8 MB
  __bf16* Xkv  = (__bf16*)(ws + 8 * MB);   // 8 MB
  __bf16* WqT  = (__bf16*)(ws + 16 * MB);  // 8 MB
  __bf16* WkvT = (__bf16*)(ws + 24 * MB);  // 16 MB
  __bf16* WoT  = (__bf16*)(ws + 40 * MB);  // 8 MB
  float*  qp   = (float*)(ws + 48 * MB);   // 16 MB
  float*  kvp  = (float*)(ws + 64 * MB);   // 32 MB
  float*  mn   = (float*)(ws + 96 * MB);   // 8 KB
  // deterministic in-stream buffer reuse (serialized by stream order):
  __bf16* qh   = (__bf16*)(ws + 0);        // reuses Xq   (dead after Q GEMM)
  __bf16* vt   = (__bf16*)(ws + 8 * MB);   // reuses Xkv  (dead after KV GEMM)
  __bf16* kh   = (__bf16*)(ws + 16 * MB);  // reuses WqT
  __bf16* attn = (__bf16*)(ws + 24 * MB);  // reuses WkvT
  float* out = (float*)d_out;

  ln_x_kernel<<<SEQ, 256, 0, stream>>>(x, qn_g, qn_b, kvn_g, kvn_b, Xq, Xkv, mn);

  dim3 tb(32, 8);
  transpose_kernel<<<dim3(DMODEL / 32, DMODEL / 32), tb, 0, stream>>>(Wq, WqT, DMODEL, DMODEL);
  transpose_kernel<<<dim3((2 * DMODEL) / 32, DMODEL / 32), tb, 0, stream>>>(Wkv, WkvT, DMODEL, 2 * DMODEL);
  transpose_kernel<<<dim3(DMODEL / 32, DMODEL / 32), tb, 0, stream>>>(Wo, WoT, DMODEL, DMODEL);

  gemm_bf16_kernel<DMODEL><<<dim3(DMODEL / 256, SEQ / 64), 256, 0, stream>>>(
      Xq, WqT, bq, qp, SEQ, DMODEL);
  gemm_bf16_kernel<DMODEL><<<dim3((2 * DMODEL) / 256, SEQ / 64), 256, 0, stream>>>(
      Xkv, WkvT, bkv, kvp, SEQ, 2 * DMODEL);

  rope_headln_kernel<<<dim3(SEQ, NHEAD), 64, 0, stream>>>(qp, kvp, mn, ln_g, ln_b, qh, kh, vt);

  flash_attn_kernel<<<(NHEAD * (SEQ / 16)) / 8, 256, 0, stream>>>(qh, kh, vt, attn);

  gemm_bf16_kernel<DMODEL><<<dim3(DMODEL / 256, SEQ / 64), 256, 0, stream>>>(
      attn, WoT, bo, out, SEQ, DMODEL);
}
